// LSTM_91293824844218
// MI455X (gfx1250) — matrix operand order repaired and checked by reading, not compiled, not run on hardware
//
#include <hip/hip_runtime.h>
#include <hip/hip_bf16.h>

// LSTM: B=128, T=512, E=H=1024.
// Phase 1: xg[t,b,:] = embeds[b,t,:] @ Wx(bf16) + bias  (WMMA GEMM, A staged
//          in LDS via async global->LDS copies, double buffered).
// Phase 2: per-t kernel: gates = xg_t + h @ Wh(bf16); both h-tile (A) and the
//          Wh slice (B) are staged in LDS with GLOBAL_LOAD_ASYNC_TO_LDS_B128,
//          s_wait_asynccnt + barrier pipelined against the WMMAs; fused
//          sigmoid/tanh + c/h update. h ping-pongs between two bf16 buffers.

typedef __bf16 bf16_t;
typedef __attribute__((ext_vector_type(16))) __bf16 v16bf;
typedef __attribute__((ext_vector_type(8)))  float  v8f;

#define B_  128
#define T_  512
#define E_  1024
#define H_  1024
#define G4  4096  // 4*H

#define A_STRIDE 40    // 32 K elems + 8 pad (80B rows: bank-conflict free)
#define B_STRIDE 264   // 256 cols + 8 pad (528B rows, 16B aligned)

__device__ __forceinline__ float sigmoidf_(float x) {
    return 1.0f / (1.0f + __expf(-x));
}

// One per-lane async 16B copy: LDS[lds_byte] = MEM[gptr]  (tracked by ASYNCcnt)
__device__ __forceinline__ void async_copy_b128(unsigned int lds_byte, const void* gptr) {
    asm volatile("global_load_async_to_lds_b128 %0, %1, off"
                 :: "v"(lds_byte), "v"(gptr) : "memory");
}
__device__ __forceinline__ void wait_async0() {
    asm volatile("s_wait_asynccnt 0x0" ::: "memory");
}
__device__ __forceinline__ unsigned int lds_off(const void* p) {
    return (unsigned int)(unsigned long long)p;  // LDS aperture: offset = addr[31:0]
}

// ---------------- pack / convert kernels ----------------

__global__ void pack_gate_bf16(const float* __restrict__ src,
                               bf16_t* __restrict__ dst, int rows, int gate) {
    int total = rows * H_;
    for (int idx = blockIdx.x * blockDim.x + threadIdx.x; idx < total;
         idx += gridDim.x * blockDim.x) {
        int r = idx >> 10;
        int c = idx & (H_ - 1);
        dst[(size_t)r * G4 + gate * H_ + c] = (bf16_t)src[idx];
    }
}

__global__ void pack_bias(const float* __restrict__ bi, const float* __restrict__ bf,
                          const float* __restrict__ bo, const float* __restrict__ bg,
                          float* __restrict__ dst) {
    int i = blockIdx.x * blockDim.x + threadIdx.x;
    if (i < H_) {
        dst[i]          = bi[i];
        dst[H_ + i]     = bf[i];
        dst[2 * H_ + i] = bo[i];
        dst[3 * H_ + i] = bg[i];
    }
}

__global__ void cvt_f32_to_bf16(const float* __restrict__ s,
                                bf16_t* __restrict__ d, long n) {
    for (long i = (long)blockIdx.x * blockDim.x + threadIdx.x; i < n;
         i += (long)gridDim.x * blockDim.x) {
        d[i] = (bf16_t)s[i];
    }
}

__global__ void init_state(bf16_t* __restrict__ h, float* __restrict__ c) {
    int i = blockIdx.x * blockDim.x + threadIdx.x;
    if (i < B_ * H_) {
        h[i] = (bf16_t)0.0f;
        c[i] = 0.0f;
    }
}

// ---------------- phase 1: xg = X @ Wx + b ----------------
// X: [B*T, E] bf16 (row m = b*T+t), Wx: [E,4H] bf16, xg: [T,B,4H] f32.
// Block = 16 rows x 512 cols (8 waves x 64). A tile staged in LDS (async,
// double buffered, shared by all 8 waves); B streamed from global/L2.
__global__ void __launch_bounds__(256)
xg_gemm(const bf16_t* __restrict__ X, const bf16_t* __restrict__ Wx,
        const float* __restrict__ bias, float* __restrict__ xg) {
    __shared__ bf16_t Ast[2][16 * A_STRIDE];

    const int tid  = threadIdx.x;
    const int lane = tid & 31;
    const int wave = tid >> 5;
    const int m0   = blockIdx.x * 16;
    const int b    = m0 >> 9;
    const int t0   = m0 & (T_ - 1);
    const int n0   = blockIdx.y * 512 + wave * 64;
    const int mrow = lane & 15;
    const int ksel = lane >> 4;

    auto stageA = [&](int buf, int k0) {
        if (tid < 64) {  // 16 rows x 4 chunks of 16B, waves 0-1 (uniform EXEC)
            int row = tid >> 2;
            int kp  = (tid & 3) * 8;
            async_copy_b128(lds_off(&Ast[buf][row * A_STRIDE + kp]),
                            X + (size_t)(m0 + row) * E_ + k0 + kp);
        }
    };

    stageA(0, 0);
    wait_async0();
    __syncthreads();

    v8f acc[4] = {};
    int buf = 0;
    for (int k0 = 0; k0 < E_; k0 += 32) {
        if (k0 + 32 < E_) stageA(buf ^ 1, k0 + 32);

        const bf16_t* ar = &Ast[buf][mrow * A_STRIDE];
        v16bf a;
        *((uint4*)&a + 0) = *(const uint4*)(ar + ksel * 8);
        *((uint4*)&a + 1) = *(const uint4*)(ar + 16 + ksel * 8);
        const bf16_t* brow = Wx + (size_t)k0 * G4 + n0 + mrow;
#pragma unroll
        for (int j = 0; j < 4; ++j) {
            v16bf bb;
#pragma unroll
            for (int s = 0; s < 16; ++s) {
                int kk = (s < 8) ? (ksel * 8 + s) : (16 + ksel * 8 + s - 8);
                bb[s] = brow[(size_t)kk * G4 + j * 16];
            }
            acc[j] = __builtin_amdgcn_wmma_f32_16x16x32_bf16(
                false, a, false, bb, (short)0, acc[j], false, false);
        }
        wait_async0();
        __syncthreads();
        buf ^= 1;
    }
#pragma unroll
    for (int j = 0; j < 4; ++j) {
        int n = n0 + j * 16 + mrow;
        float bs = bias[n];
#pragma unroll
        for (int r = 0; r < 8; ++r) {
            int t = t0 + r + ksel * 8;
            xg[((size_t)t * B_ + b) * G4 + n] = acc[j][r] + bs;
        }
    }
}

// ---------------- phase 2: one LSTM timestep ----------------
// Block = 32 batch rows x 64 H-cols x 4 gates. 8 waves: wave = (mi, ni),
// mi in {0,1} picks the 16-row half, ni in {0..3} picks 16 cols.
// A (h tile, 32x32 per K-step) and B (Wh slice, 32x256) staged in LDS
// via async copies, double buffered. grid = (4, 16).
__global__ void __launch_bounds__(256)
lstm_step(const bf16_t* __restrict__ hin, bf16_t* __restrict__ hout,
          const bf16_t* __restrict__ Wh, const float* __restrict__ xg,
          float* __restrict__ c, float* __restrict__ out, int t) {
    __shared__ bf16_t Ast[2][32 * A_STRIDE];
    __shared__ bf16_t Bst[2][32 * B_STRIDE];

    const int tid   = threadIdx.x;
    const int lane  = tid & 31;
    const int wave  = tid >> 5;
    const int mi    = wave >> 2;
    const int ni    = wave & 3;
    const int blkm0 = blockIdx.x * 32;
    const int m0    = blkm0 + mi * 16;
    const int nbase = blockIdx.y * 64;           // within H
    const int mrow  = lane & 15;
    const int ksel  = lane >> 4;

    auto stage = [&](int buf, int k0) {
        // B: 32 rows x 256 stage cols (4 gates x 64) = 1024 16B chunks
#pragma unroll
        for (int i = 0; i < 4; ++i) {
            int id  = tid + i * 256;
            int row = id >> 5;                   // 32 chunks per row
            int sc  = (id & 31) * 8;             // stage col (elems)
            int g   = sc >> 6;                   // gate
            int col = nbase + (sc & 63);
            async_copy_b128(lds_off(&Bst[buf][row * B_STRIDE + sc]),
                            Wh + (size_t)(k0 + row) * G4 + g * H_ + col);
        }
        // A: 32 rows x 32 K = 128 chunks, waves 0-3 (uniform EXEC)
        if (tid < 128) {
            int row = tid >> 2;
            int kp  = (tid & 3) * 8;
            async_copy_b128(lds_off(&Ast[buf][row * A_STRIDE + kp]),
                            hin + (size_t)(blkm0 + row) * H_ + k0 + kp);
        }
    };

    stage(0, 0);
    wait_async0();
    __syncthreads();

    v8f acc[4] = {};  // i, f, o, g
    int buf = 0;
    for (int k0 = 0; k0 < H_; k0 += 32) {
        if (k0 + 32 < H_) stage(buf ^ 1, k0 + 32);

        const bf16_t* ar = &Ast[buf][(mi * 16 + mrow) * A_STRIDE];
        v16bf a;
        *((uint4*)&a + 0) = *(const uint4*)(ar + ksel * 8);
        *((uint4*)&a + 1) = *(const uint4*)(ar + 16 + ksel * 8);
        const bf16_t* br = &Bst[buf][ni * 16 + mrow];
#pragma unroll
        for (int g = 0; g < 4; ++g) {
            v16bf bb;
#pragma unroll
            for (int s = 0; s < 16; ++s) {
                int kk = (s < 8) ? (ksel * 8 + s) : (16 + ksel * 8 + s - 8);
                bb[s] = br[kk * B_STRIDE + g * 64];
            }
            acc[g] = __builtin_amdgcn_wmma_f32_16x16x32_bf16(
                false, a, false, bb, (short)0, acc[g], false, false);
        }
        wait_async0();
        __syncthreads();
        buf ^= 1;
    }

    const int j = nbase + ni * 16 + mrow;
    const float* xgbase = xg + (size_t)t * B_ * G4;
#pragma unroll
    for (int r = 0; r < 8; ++r) {
        int m = m0 + r + ksel * 8;
        const float* xr = xgbase + (size_t)m * G4;
        float iv = sigmoidf_(acc[0][r] + xr[j]);
        float fv = sigmoidf_(acc[1][r] + xr[H_ + j]);
        float ov = sigmoidf_(acc[2][r] + xr[2 * H_ + j]);
        float gv = tanhf(acc[3][r] + xr[3 * H_ + j]);
        size_t cidx = (size_t)m * H_ + j;
        float cn = gv * iv + c[cidx] * fv;
        c[cidx] = cn;
        float hv = tanhf(cn) * ov;
        out[((size_t)m * T_ + t) * H_ + j] = hv;
        hout[cidx] = (bf16_t)hv;
    }
}

// ---------------- host side ----------------
extern "C" void kernel_launch(void* const* d_in, const int* in_sizes, int n_in,
                              void* d_out, int out_size, void* d_ws, size_t ws_size,
                              hipStream_t stream) {
    (void)in_sizes; (void)n_in; (void)out_size; (void)ws_size;
    const float* embeds = (const float*)d_in[0];
    // gate order in the reference concat: [i, f, o, g]
    const float* WxS[4] = {(const float*)d_in[1], (const float*)d_in[4],
                           (const float*)d_in[10], (const float*)d_in[7]};
    const float* WhS[4] = {(const float*)d_in[2], (const float*)d_in[5],
                           (const float*)d_in[11], (const float*)d_in[8]};
    const float* bS[4]  = {(const float*)d_in[3], (const float*)d_in[6],
                           (const float*)d_in[12], (const float*)d_in[9]};
    float* out = (float*)d_out;

    bf16_t* Wx   = (bf16_t*)d_ws;                        // E*4H bf16   (8 MB)
    bf16_t* Wh   = Wx + (size_t)E_ * G4;                 // H*4H bf16   (8 MB)
    float*  bias = (float*)(Wh + (size_t)H_ * G4);       // 4H f32
    bf16_t* h0   = (bf16_t*)(bias + G4);                 // B*H bf16
    bf16_t* h1   = h0 + (size_t)B_ * H_;                 // B*H bf16
    float*  c    = (float*)(h1 + (size_t)B_ * H_);       // B*H f32
    bf16_t* Xb   = (bf16_t*)(c + (size_t)B_ * H_);       // B*T*E bf16  (128 MB)
    float*  xg   = (float*)(Xb + (size_t)B_ * T_ * E_);  // T*B*4H f32  (1 GB)

    for (int g = 0; g < 4; ++g) {
        pack_gate_bf16<<<1024, 256, 0, stream>>>(WxS[g], Wx, E_, g);
        pack_gate_bf16<<<1024, 256, 0, stream>>>(WhS[g], Wh, H_, g);
    }
    pack_bias<<<4, 256, 0, stream>>>(bS[0], bS[1], bS[2], bS[3], bias);
    cvt_f32_to_bf16<<<8192, 256, 0, stream>>>(embeds, Xb, (long)B_ * T_ * E_);
    init_state<<<512, 256, 0, stream>>>(h0, c);

    xg_gemm<<<dim3((B_ * T_) / 16, G4 / 512), 256, 0, stream>>>(Xb, Wx, bias, xg);

    for (int t = 0; t < T_; ++t) {
        const bf16_t* hi = (t & 1) ? h1 : h0;
        bf16_t*       ho = (t & 1) ? h0 : h1;
        lstm_step<<<dim3(B_ / 32, H_ / 64), 256, 0, stream>>>(hi, ho, Wh, xg, c, out, t);
    }
}
